// Conv2dFusion_2413771620364
// MI455X (gfx1250) — matrix operand order repaired and checked by reading, not compiled, not run on hardware
//
#include <hip/hip_runtime.h>
#include <hip/hip_fp16.h>
#include <math.h>

// ---------------------------------------------------------------------------
// MI455X / gfx1250 implementation.
//   stage0: outer-product + conv0(1->32) + GELU + pool   (VALU, separable)
//   stage1: conv1(32->64) + GELU + pool   implicit GEMM, v_wmma_f32_16x16x32_f16
//   stage2: conv2(64->64) + GELU + pool   implicit GEMM, v_wmma_f32_16x16x32_f16
// f16 operands / f32 accumulate; intermediates f16 in d_ws.
// ---------------------------------------------------------------------------

typedef __attribute__((ext_vector_type(8)))  _Float16 v8h;
typedef __attribute__((ext_vector_type(16))) _Float16 v16h;
typedef __attribute__((ext_vector_type(8)))  float    v8f;

__device__ __forceinline__ float gelu_erf(float x) {
    return 0.5f * x * (1.0f + erff(x * 0.70710678118654752440f));
}

// ---------------------------------------------------------------------------
// Kernel 0: pixel(y,x) = t[y]*f[x]; conv0 + GELU + maxpool2.
// Rank-1 input => separable: F[c][dy][x] = sum_dx w[c][dy][dx]*f[x+dx]
// conv(c,y,x) = b[c] + sum_dy t[y+dy]*F[c][dy][x]   (3 FMAs per position)
// One image per block. Output f16 [512][32][63*63].
// ---------------------------------------------------------------------------
__global__ __launch_bounds__(256)
void k_outer_conv0(const float* __restrict__ token,   // [512][128]
                   const float* __restrict__ typee,   // [512][128]
                   const float* __restrict__ w0,      // [32][1][3][3]
                   const float* __restrict__ b0,      // [32]
                   _Float16*    __restrict__ buf0)    // [512][32][63*63]
{
    __shared__ float t[128];
    __shared__ float f[128];
    __shared__ float w[32 * 9];
    __shared__ float b[32];
    __shared__ float F[32 * 3 * 128];   // row-filtered input, padded rows

    const int bs  = blockIdx.x;
    const int tid = threadIdx.x;

    if (tid < 128) {
        t[tid] = typee[(size_t)bs * 128 + tid];
        f[tid] = token[(size_t)bs * 128 + tid];
    }
    for (int i = tid; i < 32 * 9; i += 256) w[i] = w0[i];
    if (tid < 32) b[tid] = b0[tid];
    __syncthreads();

    // row filter: x in [0,126)
    for (int i = tid; i < 32 * 3 * 126; i += 256) {
        const int c  = i / (3 * 126);
        const int r  = i % (3 * 126);
        const int dy = r / 126;
        const int x  = r % 126;
        const float* wc = &w[c * 9 + dy * 3];
        float v = wc[0] * f[x];
        v = fmaf(wc[1], f[x + 1], v);
        v = fmaf(wc[2], f[x + 2], v);
        F[(c * 3 + dy) * 128 + x] = v;
    }
    __syncthreads();

    // pooled output: 32 ch x 63 x 63
    for (int o = tid; o < 32 * 63 * 63; o += 256) {
        const int c   = o / (63 * 63);
        const int rem = o % (63 * 63);
        const int py  = rem / 63;
        const int px  = rem % 63;
        const float* Fc = &F[c * 3 * 128];
        const float  bc = b[c];
        float mx = -1e30f;
        #pragma unroll
        for (int q = 0; q < 4; ++q) {
            const int oy = 2 * py + (q >> 1);
            const int ox = 2 * px + (q & 1);
            float acc = bc;
            acc = fmaf(t[oy + 0], Fc[0 * 128 + ox], acc);
            acc = fmaf(t[oy + 1], Fc[1 * 128 + ox], acc);
            acc = fmaf(t[oy + 2], Fc[2 * 128 + ox], acc);
            mx  = fmaxf(mx, gelu_erf(acc));
        }
        buf0[((size_t)bs * 32 + c) * (63 * 63) + rem] = (_Float16)mx;
    }
}

// ---------------------------------------------------------------------------
// Implicit-GEMM 3x3 VALID conv + GELU + maxpool2.
//   M pooled-major: M = 4*(py*POUT+px) + (qy*2+qx); each lane's 8 acc regs =
//   8 consecutive M rows (C/D layout: VGPR r <-> row r+8*hi) => pooling is an
//   in-register max over groups of 4.
//   Block: 256 thr = 8 wave32, 4(M)x2(N): block tile 128M x 64N.
//   Wave tile: 32M x 32N -> 4 accumulators, 4 wmma per K-chunk (K-chunk = 32).
//   LDS: ping-pong A[128x32] / B[64x32] tiles + im2col offset table; the
//   weight gather is linear (w[n*KTOT + k]); the A gather is one table read.
// ---------------------------------------------------------------------------
template <int IC, int WIN, int POUT, bool FINAL>
__global__ __launch_bounds__(256)
void k_conv_wmma(const _Float16* __restrict__ in,   // [512][IC][WIN*WIN]
                 const float*    __restrict__ w,    // [64][IC][3][3] = [64][KTOT]
                 const float*    __restrict__ bias, // [64]
                 _Float16*       __restrict__ out_h,// [512][64][POUT*POUT]
                 float*          __restrict__ out_f)// [512][64*POUT*POUT]
{
    constexpr int KTOT   = IC * 9;
    static_assert(KTOT % 32 == 0, "K must be a multiple of the WMMA depth");
    constexpr int KSTEPS = KTOT / 32;
    constexpr int M      = POUT * POUT * 4;

    __shared__ __align__(16) _Float16 Alds[2][128][40];  // [buf][m][kt] (+pad)
    __shared__ __align__(16) _Float16 Blds[2][64][40];   // [buf][n][kt] (+pad)
    __shared__ int offtab[KTOT];                         // c*WIN*WIN + dy*WIN + dx

    const int bs    = blockIdx.y;
    const int mbase = blockIdx.x * 128;
    const int tid   = threadIdx.x;
    const int lane  = tid & 31;
    const int wave  = tid >> 5;
    const int wm    = wave & 3;    // M sub-tile (32 rows each)
    const int wn    = wave >> 2;   // N sub-tile (32 cols each)
    const int hi    = lane >> 4;   // fragment lane group (ISA 7.12.2)
    const int ln    = lane & 15;

    const _Float16* inb = in + (size_t)bs * IC * (WIN * WIN);
    __builtin_prefetch(w, 0, 1);   // global_prefetch_b8

    // im2col K-offset table (once per block)
    for (int k = tid; k < KTOT; k += 256) {
        const int c = k / 9, tap = k % 9;
        offtab[k] = c * (WIN * WIN) + (tap / 3) * WIN + (tap % 3);
    }

    // ---- per-thread staging coordinates: invariant across the K loop ----
    // A: 2 threads per row, 16 kt each.
    const int  arow  = tid >> 1;            // 0..127
    const int  ahalf = (tid & 1) * 16;      // kt base
    const int  gmA   = mbase + arow;
    const bool avalid = (gmA < M);
    int abase = 0;
    if (avalid) {
        const int pp = gmA >> 2, q = gmA & 3;
        const int py = pp / POUT, px = pp % POUT;
        abase = (2 * py + (q >> 1)) * WIN + (2 * px + (q & 1));
    }
    // B: 4 threads per row, 8 kt each; gather is LINEAR in k.
    const int brow = tid >> 2;              // 0..63
    const int bq   = (tid & 3) * 8;         // kt base
    const float* wrow = w + (size_t)brow * KTOT;

    __syncthreads();                        // offtab ready

    auto stage = [&](int kk, int buf) {
        const int kbase = kk * 32;
        #pragma unroll
        for (int j = 0; j < 16; ++j) {
            const int kt = ahalf + j;
            _Float16 v = (_Float16)0.0f;
            if (avalid) v = inb[abase + offtab[kbase + kt]];
            Alds[buf][arow][kt] = v;
        }
        #pragma unroll
        for (int j = 0; j < 8; ++j) {
            const int kt = bq + j;
            Blds[buf][brow][kt] = (_Float16)wrow[kbase + kt];
        }
    };
    auto ldA = [&](int buf, int m) -> v16h {
        const v8h lo = *(const v8h*)&Alds[buf][m][hi * 8];
        const v8h hv = *(const v8h*)&Alds[buf][m][16 + hi * 8];
        return __builtin_shufflevector(lo, hv, 0, 1, 2, 3, 4, 5, 6, 7,
                                       8, 9, 10, 11, 12, 13, 14, 15);
    };
    auto ldB = [&](int buf, int n) -> v16h {
        const v8h lo = *(const v8h*)&Blds[buf][n][hi * 8];
        const v8h hv = *(const v8h*)&Blds[buf][n][16 + hi * 8];
        return __builtin_shufflevector(lo, hv, 0, 1, 2, 3, 4, 5, 6, 7,
                                       8, 9, 10, 11, 12, 13, 14, 15);
    };

    v8f acc[2][2] = {};                     // [m-frag][n-frag]

    stage(0, 0);
    __syncthreads();

    for (int kk = 0; kk < KSTEPS; ++kk) {
        const int cur = kk & 1;
        if (kk + 1 < KSTEPS) stage(kk + 1, cur ^ 1);  // overlap next gather

        const int am0 = wm * 32 + ln;
        const v16h a0 = ldA(cur, am0);
        const v16h a1 = ldA(cur, am0 + 16);
        const int bn0 = wn * 32 + ln;
        const v16h b0 = ldB(cur, bn0);
        const v16h b1 = ldB(cur, bn0 + 16);

        // (neg_a, A, neg_b, B, c_mod, C, reuse_a, reuse_b)
        acc[0][0] = __builtin_amdgcn_wmma_f32_16x16x32_f16(
            false, a0, false, b0, (short)0, acc[0][0], false, false);
        acc[0][1] = __builtin_amdgcn_wmma_f32_16x16x32_f16(
            false, a0, false, b1, (short)0, acc[0][1], false, false);
        acc[1][0] = __builtin_amdgcn_wmma_f32_16x16x32_f16(
            false, a1, false, b0, (short)0, acc[1][0], false, false);
        acc[1][1] = __builtin_amdgcn_wmma_f32_16x16x32_f16(
            false, a1, false, b1, (short)0, acc[1][1], false, false);

        __syncthreads();
    }

    // ---- epilogue: bias + GELU + in-register 2x2 maxpool + store ----
    // acc[mi][ni][r] = D[row = wm*32 + mi*16 + hi*8 + r][col = wn*32 + ni*16 + ln]
    const int n0 = wn * 32 + ln;
    const float bv[2] = { bias[n0], bias[n0 + 16] };

    #pragma unroll
    for (int mi = 0; mi < 2; ++mi) {
        #pragma unroll
        for (int g = 0; g < 2; ++g) {               // two pool windows per frag
            float mx[2] = { -1e30f, -1e30f };
            #pragma unroll
            for (int r = 0; r < 4; ++r) {
                mx[0] = fmaxf(mx[0], gelu_erf(acc[mi][0][g * 4 + r] + bv[0]));
                mx[1] = fmaxf(mx[1], gelu_erf(acc[mi][1][g * 4 + r] + bv[1]));
            }
            const int gm = mbase + wm * 32 + mi * 16 + hi * 8 + g * 4;
            if (gm < M) {
                const int pp = gm >> 2;             // pooled pixel index
                if (FINAL) {
                    float* ob = out_f + (size_t)bs * (64 * POUT * POUT);
                    ob[(size_t)(n0)      * (POUT * POUT) + pp] = mx[0];
                    ob[(size_t)(n0 + 16) * (POUT * POUT) + pp] = mx[1];
                } else {
                    _Float16* ob = out_h + (size_t)bs * 64 * (POUT * POUT);
                    ob[(size_t)(n0)      * (POUT * POUT) + pp] = (_Float16)mx[0];
                    ob[(size_t)(n0 + 16) * (POUT * POUT) + pp] = (_Float16)mx[1];
                }
            }
        }
    }
}

// ---------------------------------------------------------------------------
// Launch
// ---------------------------------------------------------------------------
extern "C" void kernel_launch(void* const* d_in, const int* in_sizes, int n_in,
                              void* d_out, int out_size, void* d_ws, size_t ws_size,
                              hipStream_t stream) {
    (void)in_sizes; (void)n_in; (void)out_size; (void)ws_size;

    const float* token = (const float*)d_in[0];  // [8][64][128]
    const float* typee = (const float*)d_in[1];  // [8][64][128]
    const float* w0    = (const float*)d_in[2];  // [32][1][3][3]
    const float* b0    = (const float*)d_in[3];  // [32]
    const float* w1    = (const float*)d_in[4];  // [64][32][3][3]
    const float* b1    = (const float*)d_in[5];  // [64]
    const float* w2    = (const float*)d_in[6];  // [64][64][3][3]
    const float* b2    = (const float*)d_in[7];  // [64]
    float*       out   = (float*)d_out;          // [8][64][64*14*14]

    _Float16* buf0 = (_Float16*)d_ws;                    // [512][32][63*63] f16
    _Float16* buf1 = buf0 + (size_t)512 * 32 * 63 * 63;  // [512][64][30*30] f16

    // Stage 0: outer product + conv0 + GELU + pool (one image per block)
    k_outer_conv0<<<dim3(512), dim3(256), 0, stream>>>(token, typee, w0, b0, buf0);

    // Stage 1: conv1 32->64 over 63x63 -> pooled 30x30.  M = 3600, tiles of 128
    {
        const int mtiles = (30 * 30 * 4 + 127) / 128;    // 29
        k_conv_wmma<32, 63, 30, false>
            <<<dim3(mtiles, 512), dim3(256), 0, stream>>>(buf0, w1, b1, buf1, nullptr);
    }

    // Stage 2: conv2 64->64 over 30x30 -> pooled 14x14 -> f32 output.  M = 784
    {
        const int mtiles = (14 * 14 * 4 + 127) / 128;    // 7
        k_conv_wmma<64, 30, 14, true>
            <<<dim3(mtiles, 512), dim3(256), 0, stream>>>(buf1, w2, b2, nullptr, out);
    }
}
